// DeepReservoirMemoryNetwork_56719338111513
// MI455X (gfx1250) — matrix operand order repaired and checked by reading, not compiled, not run on hardware
//
#include <hip/hip_runtime.h>
#include <math.h>

// ---------------------------------------------------------------------------
// Deep reservoir memory network, MI455X (gfx1250) persistent-kernel design.
//   B=32, T=2048, I=64, M=H=1024, leak A=0.5
// Per step (sequential): m1 -> m2 -> h1 -> h2, each a [32x1024]x[1024x1024]^T
// GEMM group. bf16 WMMA (f32 accum), 128 WGs x 8 waves, 16x16 tiles, K/8 split
// per wave, LDS ds_add_f32 reduction, grid-wide barrier between stages.
// ---------------------------------------------------------------------------

typedef __bf16 bf16;
typedef __attribute__((ext_vector_type(16))) __bf16 v16bf;
typedef __attribute__((ext_vector_type(8)))  float  v8f;

#define B_  32
#define T_  2048
#define I_  64
#define M_  1024
#define H_  1024
#define NWG 128          // 2 (M-tiles) x 64 (N-tiles)
#define NTHR 256         // 8 wave32
#define SSTATE (B_ * M_) // 32768 elements per state buffer

union V16 { uint4 q[2]; v16bf v; };

// A fragment (16x32, bf16) from row-major activations [rows x stride].
// ISA layout: lanes 0-15 = rows 0-15 with K(k0+0..7) in V0-3, K(k0+16..23) in
// V4-7; lanes 16-31 = same rows with K(k0+8..15) / K(k0+24..31).
__device__ inline v16bf ldA_bf(const bf16* base, int stride, int k0, int lane) {
  int row = lane & 15, half = lane >> 4;
  const bf16* p = base + (size_t)row * stride + k0 + half * 8;
  V16 u;
  u.q[0] = *(const uint4*)p;        // 8 bf16 (global_load_b128)
  u.q[1] = *(const uint4*)(p + 16); // 8 bf16
  return u.v;
}

// A fragment built from strided fp32 memory (the raw input x), converted
// in-register to bf16 (only used for the tiny K=64 input GEMMs).
__device__ inline v16bf ldA_f32(const float* base, int stride, int k0, int lane) {
  int row = lane & 15, half = lane >> 4;
  const float* p0 = base + (size_t)row * stride + k0 + half * 8;
  const float* p1 = p0 + 16;
  v16bf v;
#pragma unroll
  for (int i = 0; i < 8; ++i) { v[i] = (__bf16)p0[i]; v[8 + i] = (__bf16)p1[i]; }
  return v;
}

// B fragment (32x16, bf16) = W^T tile from row-major W[N x K]:
// lane holds column n = n0+(lane&15) -> contiguous row of W, K chunks
// k0 + (lane>>4)*8 .. +8 and +16 .. +24 (same as A). Two b128 loads per lane.
__device__ inline v16bf ldB_bf(const bf16* W, int stride, int n0, int k0, int lane) {
  const bf16* p = W + (size_t)(n0 + (lane & 15)) * stride + k0 + (lane >> 4) * 8;
  V16 u;
  u.q[0] = *(const uint4*)p;
  u.q[1] = *(const uint4*)(p + 16);
  return u.v;
}

__device__ inline v8f wmma_bf16(v16bf a, v16bf b, v8f c) {
  return __builtin_amdgcn_wmma_f32_16x16x32_bf16(false, a, false, b,
                                                 (short)0, c, false, false);
}

// acc += A[16 x kb..ke] * W^T[kb..ke x 16]
__device__ inline void gemm_bf16(v8f& acc, const bf16* A, int As,
                                 const bf16* W, int Ws, int n0,
                                 int kb, int ke, int lane) {
#pragma unroll 4
  for (int k = kb; k < ke; k += 32) {
    v16bf a = ldA_bf(A, As, k, lane);
    v16bf b = ldB_bf(W, Ws, n0, k, lane);
    acc = wmma_bf16(a, b, acc);
  }
}

// Reduce 8 partial 16x16 f32 tiles (one per wave) into LDS via ds_add_f32.
// C/D layout: VGPR r of lane L holds (row = (L>=16?8:0)+r, col = L&15).
__device__ inline void reduce_tile(float* red, const v8f& acc, int tid) {
  __syncthreads();                 // previous consumers of red are done
  red[tid] = 0.0f;                 // 256 threads zero 256 floats
  __syncthreads();
  int lane = tid & 31;
  int c  = lane & 15;
  int r0 = (lane >> 4) * 8;
#pragma unroll
  for (int r = 0; r < 8; ++r)
    atomicAdd(&red[(r0 + r) * 16 + c], acc[r]);   // ds_add_f32
  __syncthreads();
}

// Device-wide sense-reversing barrier (persistent kernel, NWG co-resident WGs).
__device__ inline void grid_barrier(unsigned* cnt, volatile unsigned* gen,
                                    unsigned nwg) {
  __syncthreads();
  __threadfence();
  if (threadIdx.x == 0) {
    unsigned g = *gen;
    if (atomicAdd(cnt, 1u) == nwg - 1u) {
      *cnt = 0u;
      __threadfence();
      atomicAdd((unsigned*)gen, 1u);
    } else {
      while (*gen == g) { __builtin_amdgcn_s_sleep(2); }
    }
  }
  __syncthreads();
  __threadfence();
}

// ---------------------------------------------------------------------------

__global__ void f2bf_kernel(const float* __restrict__ src, bf16* __restrict__ dst,
                            int n) {
  int i  = blockIdx.x * blockDim.x + threadIdx.x;
  int st = gridDim.x * blockDim.x;
  for (; i < n; i += st) dst[i] = (bf16)src[i];
}

__global__ void zero_u32_kernel(unsigned* __restrict__ p, int n) {
  int i  = blockIdx.x * blockDim.x + threadIdx.x;
  int st = gridDim.x * blockDim.x;
  for (; i < n; i += st) p[i] = 0u;
}

// ---------------------------------------------------------------------------

__global__ __launch_bounds__(NTHR, 1)
void drmn_persistent(const float* __restrict__ x,
                     const float* __restrict__ b1v, const float* __restrict__ b2v,
                     float* __restrict__ out,
                     const bf16* __restrict__ Wm1,  const bf16* __restrict__ Vm1,
                     const bf16* __restrict__ Wm2,  const bf16* __restrict__ Vm2,
                     const bf16* __restrict__ Win1, const bf16* __restrict__ Wh1,
                     const bf16* __restrict__ Wmh1,
                     const bf16* __restrict__ Win2, const bf16* __restrict__ Wh2,
                     const bf16* __restrict__ Wmh2,
                     bf16* m1s, bf16* m2s, bf16* h1s, bf16* h2s,
                     float* h1f, float* h2f,
                     unsigned* cnt, unsigned* gen)
{
  __shared__ float red[256];

  const int wg   = blockIdx.x;      // 0..127
  const int mt   = wg >> 6;         // batch tile 0..1
  const int nt   = wg & 63;         // unit tile 0..63
  const int n0   = nt * 16;
  const int arow = mt * 16;         // first batch row of this tile
  const int tid  = threadIdx.x;
  const int lane = tid & 31;
  const int wave = tid >> 5;        // 0..7 (K split)
  const int kb   = wave * 128;      // 128 K per wave per 1024-K matrix

  for (int t = 0; t < T_; ++t) {
    const int rb = t & 1;
    const bf16* m1r = m1s + rb * SSTATE;  bf16* m1w = m1s + (rb ^ 1) * SSTATE;
    const bf16* m2r = m2s + rb * SSTATE;  bf16* m2w = m2s + (rb ^ 1) * SSTATE;
    const bf16* h1r = h1s + rb * SSTATE;  bf16* h1w = h1s + (rb ^ 1) * SSTATE;
    const bf16* h2r = h2s + rb * SSTATE;  bf16* h2w = h2s + (rb ^ 1) * SSTATE;
    const float* xb = x + (size_t)arow * (T_ * I_) + (size_t)t * I_;

    // ---- stage 1: m1' = m1 @ Vm1^T + x_t @ Wm1^T --------------------------
    {
      v8f acc = {};
      gemm_bf16(acc, m1r + (size_t)arow * M_, M_, Vm1, M_, n0, kb, kb + 128, lane);
      if (wave == 7) {
#pragma unroll
        for (int kk = 0; kk < I_; kk += 32)
          acc = wmma_bf16(ldA_f32(xb, T_ * I_, kk, lane),
                          ldB_bf(Wm1, I_, n0, kk, lane), acc);
      }
      reduce_tile(red, acc, tid);
      if (tid < 32) {
        int c = tid & 15, r0v = (tid >> 4) * 8;
#pragma unroll
        for (int r = 0; r < 8; ++r) {
          int m = r0v + r;
          m1w[(size_t)(arow + m) * M_ + n0 + c] = (bf16)red[m * 16 + c];
        }
      }
      grid_barrier(cnt, gen, NWG);
    }

    // ---- stage 2: m2' = m2 @ Vm2^T + m1' @ Wm2^T --------------------------
    {
      v8f acc = {};
      gemm_bf16(acc, m2r + (size_t)arow * M_, M_, Vm2, M_, n0, kb, kb + 128, lane);
      gemm_bf16(acc, m1w + (size_t)arow * M_, M_, Wm2, M_, n0, kb, kb + 128, lane);
      reduce_tile(red, acc, tid);
      if (tid < 32) {
        int c = tid & 15, r0v = (tid >> 4) * 8;
#pragma unroll
        for (int r = 0; r < 8; ++r) {
          int m = r0v + r;
          m2w[(size_t)(arow + m) * M_ + n0 + c] = (bf16)red[m * 16 + c];
        }
      }
      grid_barrier(cnt, gen, NWG);
    }

    // ---- stage 3: h1' = 0.5 h1 + 0.5 tanh(x@Win1^T + h1@Wh1^T + m2'@Wmh1^T + b1)
    {
      v8f acc = {};
      gemm_bf16(acc, h1r + (size_t)arow * H_, H_, Wh1,  H_, n0, kb, kb + 128, lane);
      gemm_bf16(acc, m2w + (size_t)arow * M_, M_, Wmh1, M_, n0, kb, kb + 128, lane);
      if (wave == 0) {
#pragma unroll
        for (int kk = 0; kk < I_; kk += 32)
          acc = wmma_bf16(ldA_f32(xb, T_ * I_, kk, lane),
                          ldB_bf(Win1, I_, n0, kk, lane), acc);
      }
      reduce_tile(red, acc, tid);
      if (tid < 32) {
        int c = tid & 15, r0v = (tid >> 4) * 8;
#pragma unroll
        for (int r = 0; r < 8; ++r) {
          int m = r0v + r;
          size_t idx = (size_t)(arow + m) * H_ + n0 + c;
          float pre = red[m * 16 + c] + b1v[n0 + c];
          float h = 0.5f * h1f[idx] + 0.5f * tanhf(pre);
          h1f[idx] = h;
          h1w[idx] = (bf16)h;
        }
      }
      grid_barrier(cnt, gen, NWG);
    }

    // ---- stage 4: h2' = 0.5 h2 + 0.5 tanh(h1'@Win2^T + h2@Wh2^T + m2'@Wmh2^T + b2)
    {
      v8f acc = {};
      gemm_bf16(acc, h1w + (size_t)arow * H_, H_, Win2, H_, n0, kb, kb + 128, lane);
      gemm_bf16(acc, h2r + (size_t)arow * H_, H_, Wh2,  H_, n0, kb, kb + 128, lane);
      gemm_bf16(acc, m2w + (size_t)arow * M_, M_, Wmh2, M_, n0, kb, kb + 128, lane);
      reduce_tile(red, acc, tid);
      if (tid < 32) {
        int c = tid & 15, r0v = (tid >> 4) * 8;
#pragma unroll
        for (int r = 0; r < 8; ++r) {
          int m = r0v + r;
          size_t idx = (size_t)(arow + m) * H_ + n0 + c;
          float pre = red[m * 16 + c] + b2v[n0 + c];
          float h = 0.5f * h2f[idx] + 0.5f * tanhf(pre);
          h2f[idx] = h;
          h2w[idx] = (bf16)h;
          out[(size_t)(arow + m) * ((size_t)T_ * H_) + (size_t)t * H_ + n0 + c] = h;
        }
      }
      // no grid barrier needed before next step's stage 1 (disjoint buffers);
      // intra-WG reuse of `red` is protected inside reduce_tile.
    }
  }
}

// ---------------------------------------------------------------------------

extern "C" void kernel_launch(void* const* d_in, const int* in_sizes, int n_in,
                              void* d_out, int out_size, void* d_ws, size_t ws_size,
                              hipStream_t stream) {
  (void)in_sizes; (void)n_in; (void)out_size; (void)ws_size;

  const float* x     = (const float*)d_in[0];
  const float* Wm1f  = (const float*)d_in[1];
  const float* Vm1f  = (const float*)d_in[2];
  const float* Wm2f  = (const float*)d_in[3];
  const float* Vm2f  = (const float*)d_in[4];
  const float* Win1f = (const float*)d_in[5];
  const float* Wh1f  = (const float*)d_in[6];
  const float* Wmh1f = (const float*)d_in[7];
  const float* b1v   = (const float*)d_in[8];
  const float* Win2f = (const float*)d_in[9];
  const float* Wh2f  = (const float*)d_in[10];
  const float* Wmh2f = (const float*)d_in[11];
  const float* b2v   = (const float*)d_in[12];
  float* out = (float*)d_out;

  char* base = (char*)d_ws;
  size_t off = 0;
  auto carve = [&](size_t bytes) -> void* {
    void* p = base + off;
    off = (off + bytes + 255) & ~(size_t)255;
    return p;
  };

  // zero-initialized region: barrier counters + all state buffers
  unsigned* cnt = (unsigned*)carve(8);           // [cnt, gen]
  unsigned* gen = cnt + 1;
  bf16* m1s = (bf16*)carve(2 * SSTATE * sizeof(bf16));
  bf16* m2s = (bf16*)carve(2 * SSTATE * sizeof(bf16));
  bf16* h1s = (bf16*)carve(2 * SSTATE * sizeof(bf16));
  bf16* h2s = (bf16*)carve(2 * SSTATE * sizeof(bf16));
  float* h1f = (float*)carve(SSTATE * sizeof(float));
  float* h2f = (float*)carve(SSTATE * sizeof(float));
  size_t zero_bytes = off;

  // bf16 weight copies (resident in L2 across all 2048 steps: ~16 MB)
  bf16* Wm1  = (bf16*)carve((size_t)M_ * I_ * sizeof(bf16));
  bf16* Vm1  = (bf16*)carve((size_t)M_ * M_ * sizeof(bf16));
  bf16* Wm2  = (bf16*)carve((size_t)M_ * M_ * sizeof(bf16));
  bf16* Vm2  = (bf16*)carve((size_t)M_ * M_ * sizeof(bf16));
  bf16* Win1 = (bf16*)carve((size_t)H_ * I_ * sizeof(bf16));
  bf16* Wh1  = (bf16*)carve((size_t)H_ * H_ * sizeof(bf16));
  bf16* Wmh1 = (bf16*)carve((size_t)H_ * M_ * sizeof(bf16));
  bf16* Win2 = (bf16*)carve((size_t)H_ * H_ * sizeof(bf16));
  bf16* Wh2  = (bf16*)carve((size_t)H_ * H_ * sizeof(bf16));
  bf16* Wmh2 = (bf16*)carve((size_t)H_ * M_ * sizeof(bf16));

  // init: zero states + barrier, convert weights to bf16 (every launch ->
  // deterministic under graph replay)
  zero_u32_kernel<<<256, 256, 0, stream>>>((unsigned*)d_ws, (int)(zero_bytes / 4));
  f2bf_kernel<<<1024, 256, 0, stream>>>(Wm1f,  Wm1,  M_ * I_);
  f2bf_kernel<<<2048, 256, 0, stream>>>(Vm1f,  Vm1,  M_ * M_);
  f2bf_kernel<<<2048, 256, 0, stream>>>(Wm2f,  Wm2,  M_ * M_);
  f2bf_kernel<<<2048, 256, 0, stream>>>(Vm2f,  Vm2,  M_ * M_);
  f2bf_kernel<<<1024, 256, 0, stream>>>(Win1f, Win1, H_ * I_);
  f2bf_kernel<<<2048, 256, 0, stream>>>(Wh1f,  Wh1,  H_ * H_);
  f2bf_kernel<<<2048, 256, 0, stream>>>(Wmh1f, Wmh1, H_ * M_);
  f2bf_kernel<<<2048, 256, 0, stream>>>(Win2f, Win2, H_ * H_);
  f2bf_kernel<<<2048, 256, 0, stream>>>(Wh2f,  Wh2,  H_ * H_);
  f2bf_kernel<<<2048, 256, 0, stream>>>(Wmh2f, Wmh2, H_ * M_);

  drmn_persistent<<<NWG, NTHR, 0, stream>>>(
      x, b1v, b2v, out,
      Wm1, Vm1, Wm2, Vm2, Win1, Wh1, Wmh1, Win2, Wh2, Wmh2,
      m1s, m2s, h1s, h2s, h1f, h2f, cnt, gen);
}
